// AttentionModel_58480274702740
// MI455X (gfx1250) — matrix operand order repaired and checked by reading, not compiled, not run on hardware
//
#include <hip/hip_runtime.h>
#include <hip/hip_bf16.h>
#include <math.h>

typedef __attribute__((ext_vector_type(16))) _Float16 v16h;
typedef __attribute__((ext_vector_type(8)))  float    v8f;

#define B_   1024
#define S_   512
#define F_   4
#define D_   32
#define H_   64
#define SM1  511   // S-1

// Dynamic LDS layout (bytes):
//   [0,      32768)  mixh   512*32 f16   (WMMA A operand, f16)
//   [32768,  98304)  valf   512*32 f32   (WMMA D tile: mix @ Wv + bv)
//   then fp32 scratch: scbuf[512], red[256], wn[128], bn[32], bqv[32], bvv[32],
//   asc[32], row511[32], qv[32], colsum[32], eo[32], ctx[32], harr[64], stats[4]
#define SMEM_BYTES (32768 + 65536 + (512 + 256 + 128 + 9 * 32 + 64 + 4) * 4)

__global__ __launch_bounds__(256) void attn_model_kernel(
    const float* __restrict__ values, const int* __restrict__ names,
    const float* __restrict__ emb_table,
    const float* __restrict__ W_num, const float* __restrict__ b_num,
    const float* __restrict__ Wq,    const float* __restrict__ bq,
    const float* __restrict__ Wv,    const float* __restrict__ bv,
    const float* __restrict__ attn_scale,
    const float* __restrict__ Wp,    const float* __restrict__ bp,
    const float* __restrict__ ln_gamma, const float* __restrict__ ln_beta,
    const float* __restrict__ Ws,
    float* __restrict__ out_score, float* __restrict__ out_emb,
    float* __restrict__ out_attn,  float* __restrict__ out_h)
{
    extern __shared__ char smem[];
    _Float16* mixh  = (_Float16*)smem;                 // 512*32 f16
    float*    valf  = (float*)(smem + 32768);          // 512*32 f32
    float*    scbuf = (float*)(smem + 98304);          // 512
    float*    red   = scbuf + 512;                     // 256
    float*    wn    = red + 256;                       // 128  (W_num 4x32)
    float*    bn    = wn + 128;                        // 32
    float*    bqv   = bn + 32;                         // 32
    float*    bvv   = bqv + 32;                        // 32
    float*    asc   = bvv + 32;                        // 32
    float*    row511= asc + 32;                        // 32  (fp32 mix last row)
    float*    qv    = row511 + 32;                     // 32
    float*    colsum= qv + 32;                         // 32
    float*    eo    = colsum + 32;                     // 32
    float*    ctx   = eo + 32;                         // 32
    float*    harr  = ctx + 32;                        // 64
    float*    stats = harr + 64;                       // 4

    const int t    = threadIdx.x;
    const int b    = blockIdx.x;
    const int lane = t & 31;
    const int wave = t >> 5;

    // ---- Preload WMMA B fragments of Wv (both 16-col N tiles), f32 -> f16.
    // B in contiguous k order (same as A): lanes 0-15 hold K=0..15 (element k = B[k][n]),
    // lanes 16-31 hold K=16..31.
    v16h bf0, bf1;
    {
        const int n  = lane & 15;
        const int kb = (lane >> 4) * 16;
        #pragma unroll
        for (int k = 0; k < 16; ++k) {
            bf0[k] = (_Float16)Wv[(kb + k) * D_ + n];
            bf1[k] = (_Float16)Wv[(kb + k) * D_ + (n + 16)];
        }
    }

    // ---- Stage small tables in LDS
    if (t < 128) wn[t] = W_num[t];
    if (t < 32) {
        bn[t]  = b_num[t];
        bqv[t] = bq[t];
        bvv[t] = bv[t];
        asc[t] = attn_scale[t];
        colsum[t] = 0.f;
        eo[t]     = 0.f;
    }
    __syncthreads();

    // ---- Phase A: mix = emb[names] + values @ W_num + b_num  (2 rows/thread)
    float csacc[32];
    #pragma unroll
    for (int d = 0; d < 32; ++d) csacc[d] = 0.f;

    #pragma unroll
    for (int rr = 0; rr < 2; ++rr) {
        const int s = t + rr * 256;
        const float4 vv  = ((const float4*)values)[(size_t)b * S_ + s];
        const int   name = names[(size_t)b * S_ + s];
        const float4* erow = (const float4*)(emb_table + (size_t)name * D_);
        _Float16* mrow = mixh + s * D_;
        #pragma unroll
        for (int g = 0; g < 8; ++g) {
            const float4 e4 = erow[g];
            const float ev[4] = {e4.x, e4.y, e4.z, e4.w};
            #pragma unroll
            for (int j = 0; j < 4; ++j) {
                const int d = g * 4 + j;
                const float m = ev[j] + bn[d]
                              + vv.x * wn[d] + vv.y * wn[32 + d]
                              + vv.z * wn[64 + d] + vv.w * wn[96 + d];
                mrow[d] = (_Float16)m;
                csacc[d] += m;
                if (s == 511) row511[d] = m;   // fp32 query row
            }
        }
    }
    #pragma unroll
    for (int d = 0; d < 32; ++d) atomicAdd(&colsum[d], csacc[d]);  // ds_add_f32
    __syncthreads();

    // ---- Phase B: q = mix[511] @ Wq + bq   (32 lanes)
    if (t < 32) {
        float acc = bqv[t];
        #pragma unroll
        for (int k = 0; k < 32; ++k) acc += row511[k] * Wq[k * D_ + t];
        qv[t] = acc;
    }

    // ---- Phase C: val = mix @ Wv + bv via v_wmma_f32_16x16x32_f16
    // 32 M-tiles of 16 rows; wave w handles tiles 4w..4w+3, both N tiles.
    {
        const int m  = lane & 15;
        const int hh = lane >> 4;
        #pragma unroll
        for (int i = 0; i < 4; ++i) {
            const int m0 = (wave * 4 + i) * 16;
            // A fragment: row m, k = 16*hh + s (contiguous, as B)
            const _Float16* arow = mixh + (m0 + m) * D_;
            v16h a;
            #pragma unroll
            for (int j = 0; j < 8; ++j) {
                const int cb = 16 * hh + 2 * j;
                a[2 * j]     = arow[cb];
                a[2 * j + 1] = arow[cb + 1];
            }
            v8f c0 = {};
            v8f c1 = {};
            c0 = __builtin_amdgcn_wmma_f32_16x16x32_f16(false, a, false, bf0,
                                                        (short)0, c0, false, false);
            c1 = __builtin_amdgcn_wmma_f32_16x16x32_f16(false, a, false, bf1,
                                                        (short)0, c1, false, false);
            // D: VGPR r holds M = 8*hh + r, N = lane&15 (+16 for tile 1)
            #pragma unroll
            for (int r = 0; r < 8; ++r) {
                const int mr = m0 + 8 * hh + r;
                valf[mr * D_ + m]      = c0[r] + bvv[m];
                valf[mr * D_ + m + 16] = c1[r] + bvv[m + 16];
            }
        }
    }
    __syncthreads();

    // ---- Phase D: scores = sum_d scale*tanh(q+val); softmax over 511
    float lmax = -1e30f;
    for (int s = t; s < SM1; s += 256) {
        const float* vr = valf + s * D_;
        float sc = 0.f;
        #pragma unroll
        for (int d = 0; d < 32; ++d) sc += asc[d] * tanhf(qv[d] + vr[d]);
        scbuf[s] = sc;
        lmax = fmaxf(lmax, sc);
    }
    red[t] = lmax;
    __syncthreads();
    for (int st = 128; st > 0; st >>= 1) {
        if (t < st) red[t] = fmaxf(red[t], red[t + st]);
        __syncthreads();
    }
    const float gmax = red[0];
    __syncthreads();

    float lsum = 0.f;
    for (int s = t; s < SM1; s += 256) {
        const float e = __expf(scbuf[s] - gmax);
        scbuf[s] = e;
        lsum += e;
    }
    red[t] = lsum;
    __syncthreads();
    for (int st = 128; st > 0; st >>= 1) {
        if (t < st) red[t] += red[t + st];
        __syncthreads();
    }
    const float inv_sum = 1.f / red[0];

    // attn output
    for (int s = t; s < SM1; s += 256)
        out_attn[(size_t)b * SM1 + s] = scbuf[s] * inv_sum;

    // emb_out[d] = sum_s attn[s] * val[s,d]
    {
        const int d = t & 31;
        float acc = 0.f;
        for (int s = (t >> 5); s < SM1; s += 8)
            acc += scbuf[s] * valf[s * D_ + d];
        atomicAdd(&eo[d], acc);
    }
    __syncthreads();

    if (t < 32) {
        const float e = eo[t] * inv_sum;
        out_emb[(size_t)b * D_ + t] = e;
        ctx[t] = e + colsum[t] * (1.f / 512.f);   // mean(emb_out + mix, axis=1)
    }
    __syncthreads();

    // ---- Phase E: h = relu(ctx @ Wp + bp); LayerNorm; softmax(h @ Ws)
    if (t < H_) {
        float acc = bp[t];
        #pragma unroll
        for (int k = 0; k < 32; ++k) acc += ctx[k] * Wp[k * H_ + t];
        harr[t] = fmaxf(acc, 0.f);
    }
    __syncthreads();
    if (t == 0) {
        float s1 = 0.f, s2 = 0.f;
        for (int i = 0; i < H_; ++i) { s1 += harr[i]; s2 += harr[i] * harr[i]; }
        const float mu  = s1 * (1.f / H_);
        const float var = s2 * (1.f / H_) - mu * mu;
        stats[0] = mu;
        stats[1] = rsqrtf(var + 1e-3f);
    }
    __syncthreads();
    if (t < H_) {
        const float hn = (harr[t] - stats[0]) * stats[1] * ln_gamma[t] + ln_beta[t];
        out_h[(size_t)b * H_ + t] = hn;
        harr[t] = hn;
    }
    __syncthreads();
    if (t == 0) {
        float l0 = 0.f, l1 = 0.f;
        for (int k = 0; k < H_; ++k) {
            l0 += harr[k] * Ws[k * 2 + 0];
            l1 += harr[k] * Ws[k * 2 + 1];
        }
        const float mx = fmaxf(l0, l1);
        const float e0 = __expf(l0 - mx), e1 = __expf(l1 - mx);
        const float inv = 1.f / (e0 + e1);
        out_score[(size_t)b * 2 + 0] = e0 * inv;
        out_score[(size_t)b * 2 + 1] = e1 * inv;
    }
}

extern "C" void kernel_launch(void* const* d_in, const int* in_sizes, int n_in,
                              void* d_out, int out_size, void* d_ws, size_t ws_size,
                              hipStream_t stream) {
    (void)in_sizes; (void)n_in; (void)out_size; (void)d_ws; (void)ws_size;

    const float* values = (const float*)d_in[0];
    const int*   names  = (const int*)  d_in[1];
    const float* emb    = (const float*)d_in[2];
    const float* W_num  = (const float*)d_in[3];
    const float* b_num  = (const float*)d_in[4];
    const float* Wq     = (const float*)d_in[5];
    const float* bq     = (const float*)d_in[6];
    const float* Wv     = (const float*)d_in[7];
    const float* bv     = (const float*)d_in[8];
    const float* ascale = (const float*)d_in[9];
    const float* Wp     = (const float*)d_in[10];
    const float* bp     = (const float*)d_in[11];
    const float* gamma  = (const float*)d_in[12];
    const float* beta   = (const float*)d_in[13];
    const float* Ws     = (const float*)d_in[14];

    float* out_score = (float*)d_out;                      // [B,2]
    float* out_emb   = out_score + (size_t)B_ * 2;         // [B,1,32]
    float* out_attn  = out_emb   + (size_t)B_ * D_;        // [B,1,511]
    float* out_h     = out_attn  + (size_t)B_ * SM1;       // [B,64]

    // Dynamic LDS > 64 KB: raise the per-kernel cap (non-stream call; safe
    // under graph capture, deterministic).
    static_assert(SMEM_BYTES <= 160 * 1024, "LDS budget");
    hipFuncSetAttribute((const void*)attn_model_kernel,
                        hipFuncAttributeMaxDynamicSharedMemorySize, SMEM_BYTES);

    attn_model_kernel<<<dim3(B_), dim3(256), SMEM_BYTES, stream>>>(
        values, names, emb, W_num, b_num, Wq, bq, Wv, bv, ascale,
        Wp, bp, gamma, beta, Ws, out_score, out_emb, out_attn, out_h);
}
